// PredictiveCodingLayer_36344013258753
// MI455X (gfx1250) — matrix operand order repaired and checked by reading, not compiled, not run on hardware
//
#include <hip/hip_runtime.h>

// CDNA5 / gfx1250 predictive-coding layer.
// - bf16 WMMA (v_wmma_f32_16x16x32_bf16) for both tied-weight GEMMs
// - double-buffered LDS tiles fed by GLOBAL_LOAD_ASYNC_TO_LDS_B128 (ASYNCcnt)
// - 128x128 block tile, 8 waves, 32x64 per wave (8 WMMA / K-chunk / wave)
// - fused LN/LIF epilogues; kWTA(spk)==spk exactly for binary spikes.

typedef __attribute__((ext_vector_type(16))) __bf16 v16bf;
typedef __attribute__((ext_vector_type(8)))  float  v8f;

#define B_    8192
#define I_    1024
#define H_    2048
#define STEPS_ 5
#define LR_    0.1f
#define BETA_  0.95122942450071400910f   // exp(-1/20)
#define EPS_LN_ 1e-5f
#define EPS_Q_  1e-5f

#define LDA_   40                        // padded LDS row (bf16 elems), 80 B
#define TBUF_  (128 * LDA_)              // elems per LDS buffer
#define TBUFB_ (TBUF_ * 2)               // bytes per LDS buffer

// Issue one per-lane async global->LDS 16B copy (GVS addressing:
// lds byte addr in VGPR, 32-bit per-lane offset in VGPR, uniform SGPR base).
#define ASYNC_LD_B128(ldsaddr, voff, sbase)                                   \
  asm volatile("global_load_async_to_lds_b128 %0, %1, %2"                     \
               :: "v"(ldsaddr), "v"(voff), "s"(sbase) : "memory")

#define WAIT_ASYNC0() asm volatile("s_wait_asynccnt 0x0" ::: "memory")

// -------------------- block reduction helper --------------------
__device__ __forceinline__ float block_reduce(float v, float* sbuf) {
  const int tid = threadIdx.x;
  sbuf[tid] = v;
  __syncthreads();
  #pragma unroll
  for (int s = 128; s > 0; s >>= 1) {
    if (tid < s) sbuf[tid] += sbuf[tid + s];
    __syncthreads();
  }
  float r = sbuf[0];
  __syncthreads();
  return r;
}

// -------------------- weight quantization --------------------
__global__ __launch_bounds__(256) void absmean_partial(const float* __restrict__ W,
                                                       float* __restrict__ part) {
  __shared__ float sbuf[256];
  const int tid = threadIdx.x;
  const int base = blockIdx.x * 1024;
  float s = 0.f;
  #pragma unroll
  for (int t = 0; t < 4; ++t) s += fabsf(W[base + tid + t * 256]);
  s = block_reduce(s, sbuf);
  if (tid == 0) part[blockIdx.x] = s;
}

__global__ __launch_bounds__(256) void absmean_final(const float* __restrict__ part,
                                                     float* __restrict__ scale) {
  __shared__ float sbuf[256];
  const int tid = threadIdx.x;
  float s = 0.f;
  #pragma unroll
  for (int t = 0; t < 8; ++t) s += part[tid + t * 256];
  s = block_reduce(s, sbuf);
  if (tid == 0) scale[0] = s / (float)(I_ * H_);
}

// Ternary quantize W -> bf16, writing both [I,H] and transposed [H,I].
__global__ __launch_bounds__(256) void quantize_w(const float* __restrict__ W,
                                                  const float* __restrict__ scale_p,
                                                  __bf16* __restrict__ wq,
                                                  __bf16* __restrict__ wqT) {
  const float sc = scale_p[0];
  const int total = I_ * H_;
  for (int idx = blockIdx.x * blockDim.x + threadIdx.x; idx < total;
       idx += gridDim.x * blockDim.x) {
    float w = W[idx];
    float q = rintf(w / (sc + EPS_Q_));           // round-half-even like jnp.round
    q = fminf(1.f, fmaxf(-1.f, q)) * sc;
    const int i = idx >> 11;                      // / H_
    const int h = idx & (H_ - 1);
    wq[idx] = (__bf16)q;
    wqT[(size_t)h * I_ + i] = (__bf16)q;
  }
}

// -------------------- layernorm of state (f32 -> bf16) --------------------
__global__ __launch_bounds__(256) void ln_rows(const float* __restrict__ x,
                                               const float* __restrict__ g,
                                               const float* __restrict__ b,
                                               __bf16* __restrict__ y,
                                               int ncols) {
  __shared__ float sbuf[256];
  const int row = blockIdx.x, tid = threadIdx.x;
  const int cnt = ncols >> 8;                     // ncols / 256 (<= 8)
  float v[8];
  float s = 0.f;
  for (int t = 0; t < cnt; ++t) {
    v[t] = x[(size_t)row * ncols + tid + t * 256];
    s += v[t];
  }
  s = block_reduce(s, sbuf);
  float sq = 0.f;
  for (int t = 0; t < cnt; ++t) sq += v[t] * v[t];
  sq = block_reduce(sq, sbuf);
  const float mu = s / (float)ncols;
  const float var = sq / (float)ncols - mu * mu;
  const float rs = rsqrtf(var + EPS_LN_);
  for (int t = 0; t < cnt; ++t) {
    const int col = tid + t * 256;
    y[(size_t)row * ncols + col] = (__bf16)((v[t] - mu) * rs * g[col] + b[col]);
  }
}

// -------------------- bf16 WMMA GEMM: C[M,N] = A[M,K] * Brm[N,K]^T (+bias[n]) ----
// A row-major [M,K] bf16, Brm row-major [N,K] bf16 (i.e. B^T stored N-major).
// 128x128 block tile, double-buffered LDS fed by async global->LDS copies.
__global__ __launch_bounds__(256) void gemm_bf16(const __bf16* __restrict__ A,
                                                 const __bf16* __restrict__ Bm,
                                                 float* __restrict__ C,
                                                 const float* __restrict__ bias,
                                                 int M, int N, int K) {
  __shared__ __align__(16) __bf16 As[2][TBUF_];
  __shared__ __align__(16) __bf16 Bs[2][TBUF_];

  const int tid   = threadIdx.x;
  const int lane  = tid & 31;
  const int wid   = tid >> 5;
  const int l15   = lane & 15;
  const int khalf = lane >> 4;
  const int waveM = wid & 3;                      // 4 waves along M: 32 rows each
  const int waveN = wid >> 2;                     // 2 waves along N: 64 cols each
  const int m0 = blockIdx.y * 128;
  const int n0 = blockIdx.x * 128;

  // Staging: A tile 128x32 = 512 16B chunks, B tile likewise; 2+2 chunks/thread.
  const int srow = tid >> 2, scol = tid & 3;
  unsigned voffA0 = (unsigned)(((m0 + srow)      * K + scol * 8) * 2);
  unsigned voffA1 = (unsigned)(((m0 + srow + 64) * K + scol * 8) * 2);
  unsigned voffB0 = (unsigned)(((n0 + srow)      * K + scol * 8) * 2);
  unsigned voffB1 = (unsigned)(((n0 + srow + 64) * K + scol * 8) * 2);
  const unsigned chunk0 = (unsigned)(((srow)      * LDA_ + scol * 8) * 2);
  const unsigned chunk1 = (unsigned)(((srow + 64) * LDA_ + scol * 8) * 2);
  // Aperture mapping: generic LDS address low 32 bits == DS byte address.
  const unsigned ldsA = (unsigned)(unsigned long long)(const void*)&As[0][0];
  const unsigned ldsB = (unsigned)(unsigned long long)(const void*)&Bs[0][0];
  const unsigned long long baseA = (unsigned long long)A;
  const unsigned long long baseB = (unsigned long long)Bm;

  // prologue: tile 0 -> buffer 0
  ASYNC_LD_B128(ldsA + chunk0, voffA0, baseA);
  ASYNC_LD_B128(ldsA + chunk1, voffA1, baseA);
  ASYNC_LD_B128(ldsB + chunk0, voffB0, baseB);
  ASYNC_LD_B128(ldsB + chunk1, voffB1, baseB);
  WAIT_ASYNC0();
  __syncthreads();

  v8f acc[2][4] = {};
  union Frag { v16bf v; uint4 q[2]; };

  const int kTiles = K >> 5;
  for (int kt = 0; kt < kTiles; ++kt) {
    const int buf = kt & 1;
    if (kt + 1 < kTiles) {
      // async-prefetch tile kt+1 into the other buffer while computing
      const unsigned bo = (unsigned)((kt + 1) & 1) * (unsigned)TBUFB_;
      voffA0 += 64; voffA1 += 64; voffB0 += 64; voffB1 += 64;
      ASYNC_LD_B128(ldsA + bo + chunk0, voffA0, baseA);
      ASYNC_LD_B128(ldsA + bo + chunk1, voffA1, baseA);
      ASYNC_LD_B128(ldsB + bo + chunk0, voffB0, baseB);
      ASYNC_LD_B128(ldsB + bo + chunk1, voffB1, baseB);
    }

    // A frag: lanes 0-15 -> K 0..7 & 16..23, lanes 16-31 -> K 8..15 & 24..31
    Frag af[2], bfr[4];
    #pragma unroll
    for (int ti = 0; ti < 2; ++ti) {
      const int ao = (waveM * 32 + ti * 16 + l15) * LDA_ + khalf * 8;
      af[ti].q[0] = *reinterpret_cast<const uint4*>(&As[buf][ao]);
      af[ti].q[1] = *reinterpret_cast<const uint4*>(&As[buf][ao + 16]);
    }
    // B frag: lanes 0-15 -> K 0..7 & 16..23, lanes 16-31 -> K 8..15 & 24..31
    #pragma unroll
    for (int tj = 0; tj < 4; ++tj) {
      const int bo2 = (waveN * 64 + tj * 16 + l15) * LDA_ + khalf * 8;
      bfr[tj].q[0] = *reinterpret_cast<const uint4*>(&Bs[buf][bo2]);
      bfr[tj].q[1] = *reinterpret_cast<const uint4*>(&Bs[buf][bo2 + 16]);
    }
    #pragma unroll
    for (int ti = 0; ti < 2; ++ti)
      #pragma unroll
      for (int tj = 0; tj < 4; ++tj)
        acc[ti][tj] = __builtin_amdgcn_wmma_f32_16x16x32_bf16(
            false, af[ti].v, false, bfr[tj].v, (short)0, acc[ti][tj], false, false);

    WAIT_ASYNC0();        // next tile resident before anyone flips buffers
    __syncthreads();
  }

  // epilogue: VGPR r -> row r (+8 for lanes 16-31); lane&15 -> column
  #pragma unroll
  for (int ti = 0; ti < 2; ++ti) {
    const int mb = m0 + waveM * 32 + ti * 16 + khalf * 8;
    #pragma unroll
    for (int tj = 0; tj < 4; ++tj) {
      const int n = n0 + waveN * 64 + tj * 16 + l15;
      const float bv = bias ? bias[n] : 0.0f;
      #pragma unroll
      for (int r = 0; r < 8; ++r)
        C[(size_t)(mb + r) * N + n] = acc[ti][tj][r] + bv;
    }
  }
}

// -------------------- generative LIF + error + LN(error) (one block per row) ----
__global__ __launch_bounds__(256) void gen_fuse(const float* __restrict__ pred_raw,
                                                float* __restrict__ gen_mem,
                                                const float* __restrict__ bottom_up,
                                                const float* __restrict__ esc_p,
                                                const float* __restrict__ g_err,
                                                const float* __restrict__ b_err,
                                                __bf16* __restrict__ nerr,
                                                int first, int last,
                                                float* __restrict__ out_err,
                                                float* __restrict__ out_pred,
                                                float* __restrict__ out_comb) {
  __shared__ float sbuf[256];
  const int b = blockIdx.x, tid = threadIdx.x;
  const float esc = esc_p[0];
  float err[4];
  float s = 0.f;
  #pragma unroll
  for (int t = 0; t < 4; ++t) {
    const int i = tid + t * 256;
    const size_t idx = (size_t)b * I_ + i;
    const float pin  = pred_raw[idx];
    const float prev = first ? 0.f : gen_mem[idx];
    const float mem  = BETA_ * prev + pin;
    const float spk  = (mem >= 1.f) ? 1.f : 0.f;
    gen_mem[idx] = (spk > 0.f) ? 0.f : mem;           // carried (reset) membrane
    const float e = (bottom_up[idx] - spk) * esc;
    err[t] = e;
    s += e;
    if (last) {
      out_pred[idx] = spk;
      out_err[idx]  = e;
      out_comb[(size_t)b * (I_ + H_) + i] = mem;      // pre-reset membrane
    }
  }
  s = block_reduce(s, sbuf);
  float sq = 0.f;
  #pragma unroll
  for (int t = 0; t < 4; ++t) sq += err[t] * err[t];
  sq = block_reduce(sq, sbuf);
  const float mu = s / (float)I_;
  const float var = sq / (float)I_ - mu * mu;
  const float rs = rsqrtf(var + EPS_LN_);
  #pragma unroll
  for (int t = 0; t < 4; ++t) {
    const int i = tid + t * 256;
    nerr[(size_t)b * I_ + i] = (__bf16)((err[t] - mu) * rs * g_err[i] + b_err[i]);
  }
}

// -------------------- inference LIF + state update (kWTA==identity) -------------
__global__ __launch_bounds__(256) void inf_fuse(const float* __restrict__ bu,
                                                float* __restrict__ inf_mem,
                                                const float* __restrict__ cur_in,
                                                float* __restrict__ cur_out,
                                                int first, int last,
                                                float* __restrict__ out_cur,
                                                float* __restrict__ out_comb) {
  const int total = B_ * H_;
  for (int idx = blockIdx.x * blockDim.x + threadIdx.x; idx < total;
       idx += gridDim.x * blockDim.x) {
    const float prev = first ? 0.f : inf_mem[idx];
    const float mem  = BETA_ * prev + bu[idx];
    const float spk  = (mem >= 1.f) ? 1.f : 0.f;
    inf_mem[idx] = (spk > 0.f) ? 0.f : mem;
    const float c = cur_in[idx] * (1.f - LR_) + spk * LR_;
    cur_out[idx] = c;
    if (last) {
      out_cur[idx] = c;
      const int b = idx >> 11;                        // / H_
      const int h = idx & (H_ - 1);
      out_comb[(size_t)b * (I_ + H_) + I_ + h] = mem; // pre-reset membrane
    }
  }
}

// -------------------- host launch --------------------
extern "C" void kernel_launch(void* const* d_in, const int* in_sizes, int n_in,
                              void* d_out, int out_size, void* d_ws, size_t ws_size,
                              hipStream_t stream) {
  const float* bottom_up = (const float*)d_in[0];   // [B,I]
  const float* top_down  = (const float*)d_in[1];   // [B,H]
  const float* W         = (const float*)d_in[2];   // [I,H]
  const float* b_gen     = (const float*)d_in[3];   // [I]
  const float* g_state   = (const float*)d_in[4];   // [H]
  const float* b_state   = (const float*)d_in[5];   // [H]
  const float* g_err     = (const float*)d_in[6];   // [I]
  const float* b_err     = (const float*)d_in[7];   // [I]
  const float* esc       = (const float*)d_in[8];   // scalar

  float* out      = (float*)d_out;
  float* out_cur  = out;                                            // [B,H]
  float* out_err  = out + (size_t)B_ * H_;                          // [B,I]
  float* out_comb = out + (size_t)B_ * H_ + (size_t)B_ * I_;        // [B,I+H]
  float* out_pred = out_comb + (size_t)B_ * (I_ + H_);              // [B,I]

  // workspace carve (256B aligned slabs)
  char* ws = (char*)d_ws;
  size_t off = 0;
  auto carve = [&](size_t bytes) -> char* {
    char* p = ws + off;
    off = (off + bytes + 255) & ~(size_t)255;
    return p;
  };
  __bf16* wq    = (__bf16*)carve((size_t)I_ * H_ * 2);
  __bf16* wqT   = (__bf16*)carve((size_t)I_ * H_ * 2);
  __bf16* xn    = (__bf16*)carve((size_t)B_ * H_ * 2);
  __bf16* nerr  = (__bf16*)carve((size_t)B_ * I_ * 2);
  float*  predr = (float*) carve((size_t)B_ * I_ * 4);
  float*  bu    = (float*) carve((size_t)B_ * H_ * 4);
  float*  genm  = (float*) carve((size_t)B_ * I_ * 4);
  float*  infm  = (float*) carve((size_t)B_ * H_ * 4);
  float*  cur   = (float*) carve((size_t)B_ * H_ * 4);
  float*  part  = (float*) carve(2048 * 4);
  float*  scale = (float*) carve(4);

  // ternary quantization of W (deterministic two-stage reduction)
  absmean_partial<<<2048, 256, 0, stream>>>(W, part);
  absmean_final<<<1, 256, 0, stream>>>(part, scale);
  quantize_w<<<2048, 256, 0, stream>>>(W, scale, wq, wqT);

  for (int step = 0; step < STEPS_; ++step) {
    const int first = (step == 0);
    const int last  = (step == STEPS_ - 1);
    const float* cur_in = first ? top_down : cur;

    // xn = LN(current) in bf16
    ln_rows<<<B_, 256, 0, stream>>>(cur_in, g_state, b_state, xn, H_);

    // pred_in = xn @ wq^T + b_gen   ([B,H]x[H,I] -> [B,I])
    gemm_bf16<<<dim3(I_ / 128, B_ / 128), 256, 0, stream>>>(
        xn, wq, predr, b_gen, B_, I_, H_);

    // generative LIF, error, LN(error) -> nerr (bf16)
    gen_fuse<<<B_, 256, 0, stream>>>(predr, genm, bottom_up, esc, g_err, b_err,
                                     nerr, first, last, out_err, out_pred, out_comb);

    // bu = nerr @ wq   ([B,I]x[I,H] -> [B,H]), B-operand = wqT row-major
    gemm_bf16<<<dim3(H_ / 128, B_ / 128), 256, 0, stream>>>(
        nerr, wqT, bu, nullptr, B_, H_, I_);

    // inference LIF + state update (kWTA is identity on binary spikes)
    inf_fuse<<<8192, 256, 0, stream>>>(bu, infm, cur_in, cur, first, last,
                                       out_cur, out_comb);
  }
}
